// RCCAModule_69655779606566
// MI455X (gfx1250) — matrix operand order repaired and checked by reading, not compiled, not run on hardware
//
#include <hip/hip_runtime.h>
#include <hip/hip_bf16.h>

// ---------------------------------------------------------------------------
// RCCA module for MI455X (gfx1250, wave32, WMMA bf16 16x16x32, f32 accum)
// conv3x3(1024->256)+relu -> 2x criss-cross attention -> conv3x3(256->1024)+relu
// Conv kernels: double-buffered LDS + global_load_async_to_lds_b128 pipeline,
// incremental staging pointers (no div/mod in hot loop), 4 WMMA tiles/wave.
// ---------------------------------------------------------------------------

typedef __attribute__((ext_vector_type(16))) __bf16 v16bf;
typedef __attribute__((ext_vector_type(8)))  __bf16 v8bf;
typedef __attribute__((ext_vector_type(8)))  float  v8f;

#define BDIM   128
#define HDIM   128
#define WDIM   128
#define HW     16384          // H*W
#define NPIX   32768          // B*H*W
#define CIN    1024
#define ICH    256            // intermediate channels
#define QCH    32             // q/k channels
#define NEGBIG (-1000000000.0f)

// ---------------------------------------------------------------------------
// WMMA fragment loaders (contiguous k = 16*(lane/16) + element for A and B, wave32).
// A tile in LDS: [M][strideK] row-major bf16 (16 x 32 per fragment).
// B tile in LDS: [N][strideK] (activation-major = B^T) bf16 (32 x 16 per frag).
// ---------------------------------------------------------------------------
__device__ __forceinline__ v16bf frag_a_ld(const __bf16* base, int strideK) {
    int lane = threadIdx.x & 31;
    int m  = lane & 15;
    int kb = (lane & 16) ? 16 : 0;
    const __bf16* p = base + m * strideK + kb;
    v16bf f;
#pragma unroll
    for (int j = 0; j < 8; ++j) { f[j] = p[j]; f[8 + j] = p[8 + j]; }
    return f;
}

__device__ __forceinline__ v16bf frag_b_ld(const __bf16* base, int strideK) {
    int lane = threadIdx.x & 31;
    int n  = lane & 15;
    int kb = (lane & 16) ? 16 : 0;
    const __bf16* p = base + n * strideK + kb;
    v16bf f;
#pragma unroll
    for (int j = 0; j < 16; ++j) f[j] = p[j];
    return f;
}

__device__ __forceinline__ v8f wmma_bf16(v16bf a, v16bf b, v8f c) {
    return __builtin_amdgcn_wmma_f32_16x16x32_bf16(
        /*neg_a=*/false, a, /*neg_b=*/false, b,
        /*c_mod=*/(short)0, c, /*reuse_a=*/false, /*reuse_b=*/false);
}

// ---------------------------------------------------------------------------
// CDNA5 async global->LDS copy (ASYNCcnt-tracked, bypasses VGPRs).
// Generic pointers to LDS carry the workgroup-relative offset in bits [31:0].
// ---------------------------------------------------------------------------
__device__ __forceinline__ unsigned lds_off(const void* p) {
    return (unsigned)(unsigned long long)p;
}

__device__ __forceinline__ void async_b128(unsigned ldsAddr, const void* gaddr) {
    asm volatile("global_load_async_to_lds_b128 %0, %1, off"
                 :: "v"(ldsAddr), "v"(gaddr)
                 : "memory");
}

__device__ __forceinline__ void wait_async0() {
    asm volatile("s_wait_asynccnt 0x0" ::: "memory");
}

// ---------------------------------------------------------------------------
// Prep kernels
// ---------------------------------------------------------------------------

// x NCHW f32 -> xb NHWC bf16   (B*H*W*CIN elements)
__global__ void k_cast_x(const float* __restrict__ x, __bf16* __restrict__ xb, int n) {
    int idx = blockIdx.x * 256 + threadIdx.x;
    if (idx >= n) return;
    int c   = idx & (CIN - 1);
    int pix = idx >> 10;                 // b*HW + h*W + w
    xb[idx] = (__bf16)x[((pix >> 14) * CIN + c) * HW + (pix & (HW - 1))];
}

// conv weight (O,I,3,3) f32 -> [tap][O][I] bf16
__global__ void k_prep_w3(const float* __restrict__ w, __bf16* __restrict__ o,
                          int O, int I, int n) {
    int idx = blockIdx.x * 256 + threadIdx.x;
    if (idx >= n) return;
    int i    = idx % I;
    int rest = idx / I;
    int om   = rest % O;
    int tap  = rest / O;
    o[idx] = (__bf16)w[(om * I + i) * 9 + tap];
}

// flat f32 -> bf16 cast (for 1x1 weights)
__global__ void k_cast_flat(const float* __restrict__ x, __bf16* __restrict__ y, int n) {
    int i = blockIdx.x * 256 + threadIdx.x;
    if (i < n) y[i] = (__bf16)x[i];
}

// ---------------------------------------------------------------------------
// conv3x3 core pattern (A and B kernels): implicit GEMM over 9 taps,
// double-buffered LDS, async staging with incremental pointers.
// Block tile: 64 (co) x 128 (pixels); 8 waves, 4 WMMA tiles per wave.
// ---------------------------------------------------------------------------

// conv3x3 A: Cin=1024 -> Cout=256, ReLU; writes feat (f32 NHWC) + featb (bf16).
// grid (Cout/64, NPIX/128), block 256.
__global__ void k_conv3x3_a(const __bf16* __restrict__ xb, const __bf16* __restrict__ wt,
                            float* __restrict__ feat, __bf16* __restrict__ featb) {
    const int Ci = CIN, Co = ICH;
    const int KSTEPS = 9 * (Ci / 32);
    __shared__ __bf16 aT[2][64 * 32];
    __shared__ __bf16 bT[2][128 * 32];
    int t = threadIdx.x, lane = t & 31, wv = t >> 5;
    int coBase = blockIdx.x * 64;
    int nBase  = blockIdx.y * 128;
    int mSub = wv & 3, nHalf = wv >> 2;
    v8f acc[4];
#pragma unroll
    for (int j = 0; j < 4; ++j) acc[j] = (v8f){0, 0, 0, 0, 0, 0, 0, 0};

    int aRow = t >> 2, aCol = (t & 3) * 8;     // A: 64 rows x 32, 8 halfs/thread
    int bRow = t >> 1, bCol = (t & 1) * 16;    // B: 128 rows x 32, 16 halfs/thread
    int n  = nBase + bRow;
    int pb = n >> 14, rem = n & (HW - 1);
    int h  = rem >> 7, w = rem & (WDIM - 1);

    // incremental staging state: coordinates of the NEXT step to stage
    const __bf16* aPtr = wt + (long)(coBase + aRow) * Ci + aCol;
    int dyN = -1, dxN = -1, k0N = 0;
    bool validN = ((unsigned)(h - 1) < (unsigned)HDIM) && ((unsigned)(w - 1) < (unsigned)WDIM);
    const __bf16* bPtr = xb + ((long)((pb << 14) + ((h - 1) << 7) + (w - 1))) * Ci + bCol;

    auto stage = [&](int buf) {
        async_b128(lds_off(&aT[buf][aRow * 32 + aCol]), aPtr);
        __bf16* dst = &bT[buf][bRow * 32 + bCol];
        if (validN) {
            async_b128(lds_off(dst),     bPtr);
            async_b128(lds_off(dst + 8), bPtr + 8);
        } else {
            v8bf z = {};
            *(v8bf*)dst       = z;
            *(v8bf*)(dst + 8) = z;
        }
    };
    auto advance = [&]() {
        aPtr += 32; bPtr += 32; k0N += 32;
        if (k0N == Ci) {                       // next tap
            k0N = 0;
            aPtr += (long)Co * Ci - Ci;
            if (++dxN > 1) { dxN = -1; ++dyN; }
            int y = h + dyN, x = w + dxN;
            validN = ((unsigned)y < (unsigned)HDIM) && ((unsigned)x < (unsigned)WDIM);
            bPtr = xb + ((long)((pb << 14) + (y << 7) + x)) * Ci + bCol;
        }
    };

    stage(0); advance();
    wait_async0();
    __syncthreads();
    for (int s = 0; s < KSTEPS; ++s) {
        int cur = s & 1;
        if (s + 1 < KSTEPS) { stage(cur ^ 1); advance(); }
        v16bf fa = frag_a_ld(&aT[cur][mSub * 16 * 32], 32);
#pragma unroll
        for (int j = 0; j < 4; ++j) {
            v16bf fb = frag_b_ld(&bT[cur][(nHalf * 64 + j * 16) * 32], 32);
            acc[j] = wmma_bf16(fa, fb, acc[j]);
        }
        wait_async0();
        __syncthreads();
    }
    int coChunk = coBase + mSub * 16 + ((lane & 16) ? 8 : 0);
#pragma unroll
    for (int j = 0; j < 4; ++j) {
        long p = nBase + nHalf * 64 + j * 16 + (lane & 15);
        v8f r; v8bf rb;
#pragma unroll
        for (int i = 0; i < 8; ++i) {
            float v = acc[j][i] > 0.f ? acc[j][i] : 0.f;
            r[i] = v; rb[i] = (__bf16)v;
        }
        *(v8f*)&feat[p * Co + coChunk]   = r;
        *(v8bf*)&featb[p * Co + coChunk] = rb;
    }
}

// conv3x3 B: 256 -> 1024, ReLU, writes d_out (NCHW f32).
// grid (1024/64, NPIX/128), block 256.
__global__ void k_conv3x3_b(const __bf16* __restrict__ featb, const __bf16* __restrict__ wt,
                            float* __restrict__ dout) {
    const int Ci = ICH, Co = CIN;
    const int KSTEPS = 9 * (Ci / 32);
    __shared__ __bf16 aT[2][64 * 32];
    __shared__ __bf16 bT[2][128 * 32];
    int t = threadIdx.x, lane = t & 31, wv = t >> 5;
    int coBase = blockIdx.x * 64;
    int nBase  = blockIdx.y * 128;
    int mSub = wv & 3, nHalf = wv >> 2;
    v8f acc[4];
#pragma unroll
    for (int j = 0; j < 4; ++j) acc[j] = (v8f){0, 0, 0, 0, 0, 0, 0, 0};

    int aRow = t >> 2, aCol = (t & 3) * 8;
    int bRow = t >> 1, bCol = (t & 1) * 16;
    int n  = nBase + bRow;
    int pb = n >> 14, rem = n & (HW - 1);
    int h  = rem >> 7, w = rem & (WDIM - 1);

    const __bf16* aPtr = wt + (long)(coBase + aRow) * Ci + aCol;
    int dyN = -1, dxN = -1, k0N = 0;
    bool validN = ((unsigned)(h - 1) < (unsigned)HDIM) && ((unsigned)(w - 1) < (unsigned)WDIM);
    const __bf16* bPtr = featb + ((long)((pb << 14) + ((h - 1) << 7) + (w - 1))) * Ci + bCol;

    auto stage = [&](int buf) {
        async_b128(lds_off(&aT[buf][aRow * 32 + aCol]), aPtr);
        __bf16* dst = &bT[buf][bRow * 32 + bCol];
        if (validN) {
            async_b128(lds_off(dst),     bPtr);
            async_b128(lds_off(dst + 8), bPtr + 8);
        } else {
            v8bf z = {};
            *(v8bf*)dst       = z;
            *(v8bf*)(dst + 8) = z;
        }
    };
    auto advance = [&]() {
        aPtr += 32; bPtr += 32; k0N += 32;
        if (k0N == Ci) {
            k0N = 0;
            aPtr += (long)Co * Ci - Ci;
            if (++dxN > 1) { dxN = -1; ++dyN; }
            int y = h + dyN, x = w + dxN;
            validN = ((unsigned)y < (unsigned)HDIM) && ((unsigned)x < (unsigned)WDIM);
            bPtr = featb + ((long)((pb << 14) + (y << 7) + x)) * Ci + bCol;
        }
    };

    stage(0); advance();
    wait_async0();
    __syncthreads();
    for (int s = 0; s < KSTEPS; ++s) {
        int cur = s & 1;
        if (s + 1 < KSTEPS) { stage(cur ^ 1); advance(); }
        v16bf fa = frag_a_ld(&aT[cur][mSub * 16 * 32], 32);
#pragma unroll
        for (int j = 0; j < 4; ++j) {
            v16bf fb = frag_b_ld(&bT[cur][(nHalf * 64 + j * 16) * 32], 32);
            acc[j] = wmma_bf16(fa, fb, acc[j]);
        }
        wait_async0();
        __syncthreads();
    }
    int coChunk = coBase + mSub * 16 + ((lane & 16) ? 8 : 0);
#pragma unroll
    for (int j = 0; j < 4; ++j) {
        int p  = nBase + nHalf * 64 + j * 16 + (lane & 15);
        int bb = p >> 14;
        int hw = p & (HW - 1);
        float* op = dout + ((long)(bb * Co + coChunk)) * HW + hw;
#pragma unroll
        for (int i = 0; i < 8; ++i) {
            float v = acc[j][i] > 0.f ? acc[j][i] : 0.f;
            op[(long)i * HW] = v;
        }
    }
}

// ---------------------------------------------------------------------------
// 1x1 conv (q/k/v): out[n][m] = sum_k wt[m][k]*in[n][k] + bias[m], bf16 out.
// K=256, async-staged LDS. grid (M/32, NPIX/128), block 256.
// ---------------------------------------------------------------------------
__global__ void k_gemm1x1(const __bf16* __restrict__ wt, const float* __restrict__ bias,
                          const __bf16* __restrict__ in, __bf16* __restrict__ out, int M) {
    const int K = ICH;
    __shared__ __bf16 aT[32 * 32];
    __shared__ __bf16 bT[128 * 32];
    int t = threadIdx.x, lane = t & 31, wv = t >> 5;
    int mBase = blockIdx.x * 32;
    int nBase = blockIdx.y * 128;
    int mSub = wv & 1, nQuad = wv >> 1;
    v8f acc0 = {0,0,0,0,0,0,0,0}, acc1 = {0,0,0,0,0,0,0,0};

    int aRow = t >> 2, aCol = (t & 3) * 8;   // first 128 threads stage A
    int bRow = t >> 1, bCol = (t & 1) * 16;

    for (int k0 = 0; k0 < K; k0 += 32) {
        if (t < 128)
            async_b128(lds_off(&aT[aRow * 32 + aCol]), &wt[(mBase + aRow) * K + k0 + aCol]);
        const __bf16* src = in + (long)(nBase + bRow) * K + k0 + bCol;
        async_b128(lds_off(&bT[bRow * 32 + bCol]),     src);
        async_b128(lds_off(&bT[bRow * 32 + bCol + 8]), src + 8);
        wait_async0();
        __syncthreads();
        v16bf fa  = frag_a_ld(&aT[mSub * 16 * 32], 32);
        v16bf fb0 = frag_b_ld(&bT[(nQuad * 32) * 32], 32);
        v16bf fb1 = frag_b_ld(&bT[(nQuad * 32 + 16) * 32], 32);
        acc0 = wmma_bf16(fa, fb0, acc0);
        acc1 = wmma_bf16(fa, fb1, acc1);
        __syncthreads();
    }
    int mChunk = mBase + mSub * 16 + ((lane & 16) ? 8 : 0);
    int p0 = nBase + nQuad * 32 + (lane & 15);
    float bs[8];
#pragma unroll
    for (int i = 0; i < 8; ++i) bs[i] = bias[mChunk + i];
#pragma unroll
    for (int j = 0; j < 2; ++j) {
        v8f a = j ? acc1 : acc0;
        long p = p0 + j * 16;
        v8bf o;
#pragma unroll
        for (int i = 0; i < 8; ++i) o[i] = (__bf16)(a[i] + bs[i]);
        *(v8bf*)&out[p * M + mChunk] = o;
    }
}

// ---------------------------------------------------------------------------
// Row energies: per (b,h): E[w][v] = sum_c q[c,w]*k[c,v].  K=32, single step.
// energy layout: [pixel][256] f32, row scores at [0..127].
// grid (2, 2, B*H), block 256.
// ---------------------------------------------------------------------------
__global__ void k_energy_row(const __bf16* __restrict__ q, const __bf16* __restrict__ k,
                             float* __restrict__ e) {
    __shared__ __bf16 aT[64 * 32];
    __shared__ __bf16 bT[64 * 32];
    int t = threadIdx.x, lane = t & 31, wv = t >> 5;
    int wBase = blockIdx.x * 64;
    int vBase = blockIdx.y * 64;
    long bh = blockIdx.z;
    int row = t >> 2, col8 = (t & 3) * 8;
    async_b128(lds_off(&aT[row * 32 + col8]), &q[(bh * WDIM + wBase + row) * QCH + col8]);
    async_b128(lds_off(&bT[row * 32 + col8]), &k[(bh * WDIM + vBase + row) * QCH + col8]);
    wait_async0();
    __syncthreads();
    int mSub = wv & 3, nHalf = wv >> 2;
    v16bf fa  = frag_a_ld(&aT[mSub * 16 * 32], 32);
    v16bf fb0 = frag_b_ld(&bT[(nHalf * 32) * 32], 32);
    v16bf fb1 = frag_b_ld(&bT[(nHalf * 32 + 16) * 32], 32);
    v8f z = {0,0,0,0,0,0,0,0};
    v8f d0 = wmma_bf16(fa, fb0, z);
    v8f d1 = wmma_bf16(fa, fb1, z);
    int wRow = wBase + mSub * 16 + ((lane & 16) ? 8 : 0);
    int v0 = vBase + nHalf * 32 + (lane & 15);
#pragma unroll
    for (int j = 0; j < 2; ++j) {
        v8f d = j ? d1 : d0;
#pragma unroll
        for (int i = 0; i < 8; ++i)
            e[(bh * WDIM + wRow + i) * 256 + v0 + j * 16] = d[i];
    }
}

// ---------------------------------------------------------------------------
// Column energies: per (b,w): E[h][u] = sum_c q[c,h]*k[c,u]; diag -> NEG.
// Scores at [128..255] of each pixel. grid (2, 2, B*W), block 256.
// ---------------------------------------------------------------------------
__global__ void k_energy_col(const __bf16* __restrict__ q, const __bf16* __restrict__ k,
                             float* __restrict__ e) {
    __shared__ __bf16 aT[64 * 32];
    __shared__ __bf16 bT[64 * 32];
    int t = threadIdx.x, lane = t & 31, wv = t >> 5;
    int hBase = blockIdx.x * 64;
    int uBase = blockIdx.y * 64;
    long zid = blockIdx.z;
    long b = zid >> 7;
    int  w = (int)(zid & (WDIM - 1));
    int row = t >> 2, col8 = (t & 3) * 8;
    async_b128(lds_off(&aT[row * 32 + col8]),
               &q[((b * HDIM + hBase + row) * WDIM + w) * QCH + col8]);
    async_b128(lds_off(&bT[row * 32 + col8]),
               &k[((b * HDIM + uBase + row) * WDIM + w) * QCH + col8]);
    wait_async0();
    __syncthreads();
    int mSub = wv & 3, nHalf = wv >> 2;
    v16bf fa  = frag_a_ld(&aT[mSub * 16 * 32], 32);
    v16bf fb0 = frag_b_ld(&bT[(nHalf * 32) * 32], 32);
    v16bf fb1 = frag_b_ld(&bT[(nHalf * 32 + 16) * 32], 32);
    v8f z = {0,0,0,0,0,0,0,0};
    v8f d0 = wmma_bf16(fa, fb0, z);
    v8f d1 = wmma_bf16(fa, fb1, z);
    int hRow = hBase + mSub * 16 + ((lane & 16) ? 8 : 0);
    int u0 = uBase + nHalf * 32 + (lane & 15);
#pragma unroll
    for (int j = 0; j < 2; ++j) {
        v8f d = j ? d1 : d0;
        int u = u0 + j * 16;
#pragma unroll
        for (int i = 0; i < 8; ++i) {
            int hh = hRow + i;
            float val = (hh == u) ? NEGBIG : d[i];
            e[((b * HDIM + hh) * WDIM + w) * 256 + 128 + u] = val;
        }
    }
}

// ---------------------------------------------------------------------------
// Softmax over the 256 scores of each pixel; one wave32 per pixel.
// Writes attention weights as bf16. grid NPIX/8, block 256 (8 waves).
// ---------------------------------------------------------------------------
__global__ void k_softmax(const float* __restrict__ e, __bf16* __restrict__ attn) {
    int wv = threadIdx.x >> 5, lane = threadIdx.x & 31;
    long p = (long)blockIdx.x * 8 + wv;
    const float* ep = e + p * 256 + lane * 8;
    float v[8];
#pragma unroll
    for (int i = 0; i < 8; ++i) v[i] = ep[i];
    float m = v[0];
#pragma unroll
    for (int i = 1; i < 8; ++i) m = fmaxf(m, v[i]);
#pragma unroll
    for (int off = 16; off > 0; off >>= 1) m = fmaxf(m, __shfl_xor(m, off, 32));
    float s = 0.f;
#pragma unroll
    for (int i = 0; i < 8; ++i) { v[i] = __expf(v[i] - m); s += v[i]; }
#pragma unroll
    for (int off = 16; off > 0; off >>= 1) s += __shfl_xor(s, off, 32);
    float inv = 1.0f / s;
    v8bf o;
#pragma unroll
    for (int i = 0; i < 8; ++i) o[i] = (__bf16)(v[i] * inv);
    *(v8bf*)(attn + p * 256 + lane * 8) = o;
}

// ---------------------------------------------------------------------------
// Row aggregation: per (b,h): Out[w][c] = sum_v A_row[w][v] * V[v][c]. K=128.
// grid (ICH/64, WDIM/64, B*H), block 256.
// ---------------------------------------------------------------------------
__global__ void k_agg_row(const __bf16* __restrict__ attn, const __bf16* __restrict__ vv,
                          float* __restrict__ agg) {
    __shared__ __bf16 aT[64 * 32];
    __shared__ __bf16 bT[64 * 32];
    int t = threadIdx.x, lane = t & 31, wv = t >> 5;
    int cBase = blockIdx.x * 64;
    int wBase = blockIdx.y * 64;
    long bh = blockIdx.z;
    int aRow = t >> 2, aCol = (t & 3) * 8;
    int vL = t >> 3, cL0 = (t & 7) * 8;
    int mSub = wv & 3, nHalf = wv >> 2;
    v8f acc0 = {0,0,0,0,0,0,0,0}, acc1 = {0,0,0,0,0,0,0,0};
    for (int k0 = 0; k0 < WDIM; k0 += 32) {
        async_b128(lds_off(&aT[aRow * 32 + aCol]),
                   &attn[(bh * WDIM + wBase + aRow) * 256 + k0 + aCol]);
        v8bf src = *(const v8bf*)&vv[(bh * WDIM + k0 + vL) * ICH + cBase + cL0];
#pragma unroll
        for (int i = 0; i < 8; ++i) bT[(cL0 + i) * 32 + vL] = src[i];  // transpose stage
        wait_async0();
        __syncthreads();
        v16bf fa  = frag_a_ld(&aT[mSub * 16 * 32], 32);
        v16bf fb0 = frag_b_ld(&bT[(nHalf * 32) * 32], 32);
        v16bf fb1 = frag_b_ld(&bT[(nHalf * 32 + 16) * 32], 32);
        acc0 = wmma_bf16(fa, fb0, acc0);
        acc1 = wmma_bf16(fa, fb1, acc1);
        __syncthreads();
    }
    int wRow = wBase + mSub * 16 + ((lane & 16) ? 8 : 0);
    int c0 = cBase + nHalf * 32 + (lane & 15);
#pragma unroll
    for (int j = 0; j < 2; ++j) {
        v8f a = j ? acc1 : acc0;
#pragma unroll
        for (int i = 0; i < 8; ++i)
            agg[(bh * WDIM + wRow + i) * ICH + c0 + j * 16] = a[i];
    }
}

// ---------------------------------------------------------------------------
// Column aggregation: per (b,w): Out[h][c] = sum_u A_col[h][u] * V[u][c]. K=128.
// grid (ICH/64, HDIM/64, B*W), block 256.
// ---------------------------------------------------------------------------
__global__ void k_agg_col(const __bf16* __restrict__ attn, const __bf16* __restrict__ vv,
                          float* __restrict__ agg) {
    __shared__ __bf16 aT[64 * 32];
    __shared__ __bf16 bT[64 * 32];
    int t = threadIdx.x, lane = t & 31, wv = t >> 5;
    int cBase = blockIdx.x * 64;
    int hBase = blockIdx.y * 64;
    long zid = blockIdx.z;
    long b = zid >> 7;
    int  w = (int)(zid & (WDIM - 1));
    int aRow = t >> 2, aCol = (t & 3) * 8;
    int uL = t >> 3, cL0 = (t & 7) * 8;
    int mSub = wv & 3, nHalf = wv >> 2;
    v8f acc0 = {0,0,0,0,0,0,0,0}, acc1 = {0,0,0,0,0,0,0,0};
    for (int k0 = 0; k0 < HDIM; k0 += 32) {
        async_b128(lds_off(&aT[aRow * 32 + aCol]),
                   &attn[((b * HDIM + hBase + aRow) * WDIM + w) * 256 + 128 + k0 + aCol]);
        v8bf src = *(const v8bf*)&vv[((b * HDIM + k0 + uL) * WDIM + w) * ICH + cBase + cL0];
#pragma unroll
        for (int i = 0; i < 8; ++i) bT[(cL0 + i) * 32 + uL] = src[i];
        wait_async0();
        __syncthreads();
        v16bf fa  = frag_a_ld(&aT[mSub * 16 * 32], 32);
        v16bf fb0 = frag_b_ld(&bT[(nHalf * 32) * 32], 32);
        v16bf fb1 = frag_b_ld(&bT[(nHalf * 32 + 16) * 32], 32);
        acc0 = wmma_bf16(fa, fb0, acc0);
        acc1 = wmma_bf16(fa, fb1, acc1);
        __syncthreads();
    }
    int hRow = hBase + mSub * 16 + ((lane & 16) ? 8 : 0);
    int c0 = cBase + nHalf * 32 + (lane & 15);
#pragma unroll
    for (int j = 0; j < 2; ++j) {
        v8f a = j ? acc1 : acc0;
#pragma unroll
        for (int i = 0; i < 8; ++i)
            agg[((b * HDIM + hRow + i) * WDIM + w) * ICH + c0 + j * 16] = a[i];
    }
}

// ---------------------------------------------------------------------------
// combine: feat = gamma*(aggR+aggC) + feat ; featb = bf16(feat)
// ---------------------------------------------------------------------------
__global__ void k_combine(const float* __restrict__ aggR, const float* __restrict__ aggC,
                          const float* __restrict__ gamma,
                          float* __restrict__ feat, __bf16* __restrict__ featb, int n) {
    int i = blockIdx.x * 256 + threadIdx.x;
    if (i >= n) return;
    float f = gamma[0] * (aggR[i] + aggC[i]) + feat[i];
    feat[i]  = f;
    featb[i] = (__bf16)f;
}

// ---------------------------------------------------------------------------
// Host launcher
// ---------------------------------------------------------------------------
extern "C" void kernel_launch(void* const* d_in, const int* in_sizes, int n_in,
                              void* d_out, int out_size, void* d_ws, size_t ws_size,
                              hipStream_t stream) {
    const float* x       = (const float*)d_in[0];
    const float* conva_w = (const float*)d_in[1];
    const float* qw      = (const float*)d_in[2];
    const float* qb      = (const float*)d_in[3];
    const float* kw      = (const float*)d_in[4];
    const float* kb      = (const float*)d_in[5];
    const float* vw      = (const float*)d_in[6];
    const float* vb      = (const float*)d_in[7];
    const float* gamma   = (const float*)d_in[8];
    const float* convb_w = (const float*)d_in[9];
    float* dout = (float*)d_out;

    char* ws = (char*)d_ws;
    size_t off = 0;
    auto take = [&](size_t bytes) -> char* {
        char* p = ws + off;
        off += (bytes + 255) & ~(size_t)255;
        return p;
    };
    __bf16* xb     = (__bf16*)take((size_t)NPIX * CIN * 2);      // 64 MiB NHWC bf16
    __bf16* wa     = (__bf16*)take((size_t)9 * ICH * CIN * 2);   // conv A weights
    __bf16* wb     = (__bf16*)take((size_t)9 * CIN * ICH * 2);   // conv B weights
    __bf16* qwb    = (__bf16*)take((size_t)QCH * ICH * 2);
    __bf16* kwb    = (__bf16*)take((size_t)QCH * ICH * 2);
    __bf16* vwb    = (__bf16*)take((size_t)ICH * ICH * 2);
    float*  feat   = (float*)take((size_t)NPIX * ICH * 4);       // f32 NHWC
    __bf16* featb  = (__bf16*)take((size_t)NPIX * ICH * 2);
    __bf16* qbuf   = (__bf16*)take((size_t)NPIX * QCH * 2);
    __bf16* kbuf   = (__bf16*)take((size_t)NPIX * QCH * 2);
    __bf16* vbuf   = (__bf16*)take((size_t)NPIX * ICH * 2);
    float*  energy = (float*)take((size_t)NPIX * 256 * 4);       // also reused as aggR
    __bf16* attn   = (__bf16*)take((size_t)NPIX * 256 * 2);
    float*  aggC   = (float*)take((size_t)NPIX * ICH * 4);
    float*  aggR   = energy;  // safe: softmax fully consumes energy before agg_row

    // --- prep ---
    {
        int n = NPIX * CIN;
        k_cast_x<<<(n + 255) / 256, 256, 0, stream>>>(x, xb, n);
    }
    {
        int n = 9 * ICH * CIN;
        k_prep_w3<<<(n + 255) / 256, 256, 0, stream>>>(conva_w, wa, ICH, CIN, n);
        k_prep_w3<<<(n + 255) / 256, 256, 0, stream>>>(convb_w, wb, CIN, ICH, n);
    }
    k_cast_flat<<<(QCH * ICH + 255) / 256, 256, 0, stream>>>(qw, qwb, QCH * ICH);
    k_cast_flat<<<(QCH * ICH + 255) / 256, 256, 0, stream>>>(kw, kwb, QCH * ICH);
    k_cast_flat<<<(ICH * ICH + 255) / 256, 256, 0, stream>>>(vw, vwb, ICH * ICH);

    // --- conv3x3 A (1024 -> 256) + relu ---
    k_conv3x3_a<<<dim3(ICH / 64, NPIX / 128), 256, 0, stream>>>(xb, wa, feat, featb);

    // --- 2 recurrences of criss-cross attention ---
    for (int r = 0; r < 2; ++r) {
        k_gemm1x1<<<dim3(QCH / 32, NPIX / 128), 256, 0, stream>>>(qwb, qb, featb, qbuf, QCH);
        k_gemm1x1<<<dim3(QCH / 32, NPIX / 128), 256, 0, stream>>>(kwb, kb, featb, kbuf, QCH);
        k_gemm1x1<<<dim3(ICH / 32, NPIX / 128), 256, 0, stream>>>(vwb, vb, featb, vbuf, ICH);

        k_energy_row<<<dim3(2, 2, BDIM * HDIM), 256, 0, stream>>>(qbuf, kbuf, energy);
        k_energy_col<<<dim3(2, 2, BDIM * WDIM), 256, 0, stream>>>(qbuf, kbuf, energy);

        k_softmax<<<NPIX / 8, 256, 0, stream>>>(energy, attn);

        k_agg_row<<<dim3(ICH / 64, WDIM / 64, BDIM * HDIM), 256, 0, stream>>>(attn, vbuf, aggR);
        k_agg_col<<<dim3(ICH / 64, HDIM / 64, BDIM * WDIM), 256, 0, stream>>>(attn, vbuf, aggC);

        {
            int n = NPIX * ICH;
            k_combine<<<(n + 255) / 256, 256, 0, stream>>>(aggR, aggC, gamma, feat, featb, n);
        }
    }

    // --- conv3x3 B (256 -> 1024) + relu -> d_out (NCHW f32) ---
    k_conv3x3_b<<<dim3(CIN / 64, NPIX / 128), 256, 0, stream>>>(featb, wb, dout);

    (void)in_sizes; (void)n_in; (void)out_size; (void)ws_size;
}
